// WaveRNN_80083960201384
// MI455X (gfx1250) — matrix operand order repaired and checked by reading, not compiled, not run on hardware
//
#include <hip/hip_runtime.h>
#include <hip/hip_bf16.h>

// ---------------------------------------------------------------------------
// WaveRNN GRU step + dual output MLP for MI455X (gfx1250, wave32, WMMA).
// All GEMMs: V_WMMA_F32_16X16X32_BF16, register double-buffered k-loops,
// 32-row (gate) / 32x32 (MLP) wave tiles. Addressing: uniform SGPR base +
// 32-bit per-lane BYTE offset (saddr form), strength-reduced by += 64.
// ---------------------------------------------------------------------------

typedef __attribute__((ext_vector_type(16))) __bf16 v16bf;
typedef __attribute__((ext_vector_type(8)))  float  v8f;
typedef __attribute__((ext_vector_type(4)))  unsigned int   u32x4;
typedef __attribute__((ext_vector_type(4)))  float          f32x4;
typedef __attribute__((ext_vector_type(4)))  unsigned short u16x4;

#define HIDDEN 896
#define SPLIT  448
#define QUANT  256
#define BATCH  8192

#define WMMA_BF16(A, B, C) \
    __builtin_amdgcn_wmma_f32_16x16x32_bf16(false, (A), false, (B), (short)0, (C), false, false)

// ---- fp32 -> bf16 round-to-nearest-even ----
__device__ __forceinline__ unsigned short f32_to_bf16(float f) {
    unsigned int u = __float_as_uint(f);
    u += 0x7FFFu + ((u >> 16) & 1u);
    return (unsigned short)(u >> 16);
}

union FragU { u32x4 q[2]; v16bf v; };

// A fragment: byte offset = ((rowBase + lane%16)*lda + k + (lane>=16?8:0))*2.
// Chunks at +0 and +32 bytes (K and K+16 in elements).
__device__ __forceinline__ v16bf load_a_byte(const unsigned short* __restrict__ A,
                                             unsigned offB) {
    const char* p = (const char*)A + offB;
    FragU f;
    f.q[0] = *(const u32x4*)(p);
    f.q[1] = *(const u32x4*)(p + 32);
    return f.v;
}

// B fragment: byte offset = ((colBase + lane%16)*ldw + k + (lane>=16?8:0))*2.
// Chunks at +0 and +32 bytes (K and K+16 in elements).
__device__ __forceinline__ v16bf load_b_byte(const unsigned short* __restrict__ W,
                                             unsigned offB) {
    const char* p = (const char*)W + offB;
    FragU f;
    f.q[0] = *(const u32x4*)(p);
    f.q[1] = *(const u32x4*)(p + 32);
    return f.v;
}

// Byte-offset helpers.
__device__ __forceinline__ unsigned a_boff(int rowBase, int lane, int lda) {
    return ((unsigned)(rowBase + (lane & 15)) * lda + ((lane >> 4) ? 8u : 0u)) * 2u;
}
__device__ __forceinline__ unsigned b_boff(int colBase, int lane, int ldw) {
    return ((unsigned)(colBase + (lane & 15)) * ldw + ((lane >> 4) ? 8u : 0u)) * 2u;
}

// ---------------------------------------------------------------------------
// Kernel 1: f32 -> bf16 conversion, 4 elements / thread
// ---------------------------------------------------------------------------
__global__ void cvt_bf16_kernel(const float* __restrict__ src,
                                unsigned short* __restrict__ dst, int n4) {
    int i = blockIdx.x * blockDim.x + threadIdx.x;
    if (i < n4) {
        f32x4 v = *(const f32x4*)(src + (size_t)i * 4);
        u16x4 o;
        o.x = f32_to_bf16(v.x); o.y = f32_to_bf16(v.y);
        o.z = f32_to_bf16(v.z); o.w = f32_to_bf16(v.w);
        *(u16x4*)(dst + (size_t)i * 4) = o;
    }
}

// ---------------------------------------------------------------------------
// Kernel 2: fused recurrent GEMMs + GRU gate math.
// Wave tile: 32 rows x 16 cols, 3 weight matrices -> 6 accumulators.
// grid = (HIDDEN/16, BATCH/256), block = 256 (8 waves).
// ---------------------------------------------------------------------------
__global__ __launch_bounds__(256) void gate_kernel(
    const unsigned short* __restrict__ hbf,   // bf16 prev_hidden [B,H]
    const unsigned short* __restrict__ Wru,   // bf16 [H,H]
    const unsigned short* __restrict__ Wrr,
    const unsigned short* __restrict__ Wre,
    const float* __restrict__ prev_hidden,    // f32 [B,H]
    const float* __restrict__ prev_y,         // f32 [B,2]
    const float* __restrict__ coarse,         // f32 [B,1]
    const float* __restrict__ W_Ic,           // f32 [3*SPLIT,2]
    const float* __restrict__ W_If,           // f32 [3*SPLIT,3]
    const float* __restrict__ bias_u,
    const float* __restrict__ bias_r,
    const float* __restrict__ bias_e,
    float* __restrict__ hidden_out,           // f32 [B,H] (into d_out)
    unsigned short* __restrict__ hnew_bf)     // bf16 [B,H] (workspace)
{
    const int lane    = threadIdx.x & 31;
    const int wave    = threadIdx.x >> 5;
    const int colBase = blockIdx.x * 16;
    const int rowBase = (blockIdx.y * 8 + wave) * 32;

    v8f aU0 = {}, aU1 = {}, aR0 = {}, aR1 = {}, aE0 = {}, aE1 = {};

    // 32-bit byte offsets (saddr addressing), advanced by 64 B per k-step
    unsigned offA0 = a_boff(rowBase,      lane, HIDDEN);
    unsigned offA1 = a_boff(rowBase + 16, lane, HIDDEN);
    unsigned offB  = b_boff(colBase,      lane, HIDDEN);

    // prologue loads (k = 0)
    v16bf a0 = load_a_byte(hbf, offA0);
    v16bf a1 = load_a_byte(hbf, offA1);
    v16bf bu = load_b_byte(Wru, offB);
    v16bf br = load_b_byte(Wrr, offB);
    v16bf be = load_b_byte(Wre, offB);

    for (int it = 0; it < HIDDEN / 32 - 1; ++it) {
        offA0 += 64; offA1 += 64; offB += 64;
        // issue next-iteration loads first (stay outstanding during WMMAs)
        v16bf na0 = load_a_byte(hbf, offA0);
        v16bf na1 = load_a_byte(hbf, offA1);
        v16bf nbu = load_b_byte(Wru, offB);
        v16bf nbr = load_b_byte(Wrr, offB);
        v16bf nbe = load_b_byte(Wre, offB);

        aU0 = WMMA_BF16(a0, bu, aU0);
        aU1 = WMMA_BF16(a1, bu, aU1);
        aR0 = WMMA_BF16(a0, br, aR0);
        aR1 = WMMA_BF16(a1, br, aR1);
        aE0 = WMMA_BF16(a0, be, aE0);
        aE1 = WMMA_BF16(a1, be, aE1);

        a0 = na0; a1 = na1; bu = nbu; br = nbr; be = nbe;
    }
    aU0 = WMMA_BF16(a0, bu, aU0);
    aU1 = WMMA_BF16(a1, bu, aU1);
    aR0 = WMMA_BF16(a0, br, aR0);
    aR1 = WMMA_BF16(a1, br, aR1);
    aE0 = WMMA_BF16(a0, be, aE0);
    aE1 = WMMA_BF16(a1, be, aE1);

    // ----- epilogue: tiny input projections + gate math in f32 -----
    const int col = colBase + (lane & 15);   // fixed hidden index for this lane
    float wu0, wu1, wu2, wr0, wr1, wr2, we0, we1, we2;
    if (col < SPLIT) {                        // coarse half: I = prev_y @ W_Ic^T
        wu0 = W_Ic[(size_t)col * 2];               wu1 = W_Ic[(size_t)col * 2 + 1];               wu2 = 0.f;
        wr0 = W_Ic[(size_t)(SPLIT + col) * 2];     wr1 = W_Ic[(size_t)(SPLIT + col) * 2 + 1];     wr2 = 0.f;
        we0 = W_Ic[(size_t)(2 * SPLIT + col) * 2]; we1 = W_Ic[(size_t)(2 * SPLIT + col) * 2 + 1]; we2 = 0.f;
    } else {                                  // fine half: I = [prev_y, coarse] @ W_If^T
        const int hf = col - SPLIT;
        wu0 = W_If[(size_t)hf * 3];               wu1 = W_If[(size_t)hf * 3 + 1];               wu2 = W_If[(size_t)hf * 3 + 2];
        wr0 = W_If[(size_t)(SPLIT + hf) * 3];     wr1 = W_If[(size_t)(SPLIT + hf) * 3 + 1];     wr2 = W_If[(size_t)(SPLIT + hf) * 3 + 2];
        we0 = W_If[(size_t)(2 * SPLIT + hf) * 3]; we1 = W_If[(size_t)(2 * SPLIT + hf) * 3 + 1]; we2 = W_If[(size_t)(2 * SPLIT + hf) * 3 + 2];
    }
    const float bu_ = bias_u[col], br_ = bias_r[col], be_ = bias_e[col];
    const int hi = (lane >> 4) ? 8 : 0;

    auto epi = [&](const v8f& aU, const v8f& aR, const v8f& aE, int rb) {
#pragma unroll
        for (int r = 0; r < 8; ++r) {
            const int row = rb + r + hi;
            const float py0 = prev_y[(size_t)row * 2];
            const float py1 = prev_y[(size_t)row * 2 + 1];
            const float cc  = coarse[row];
            const float Iu = py0 * wu0 + py1 * wu1 + cc * wu2;
            const float Ir = py0 * wr0 + py1 * wr1 + cc * wr2;
            const float Ie = py0 * we0 + py1 * we1 + cc * we2;
            const float u  = 1.0f / (1.0f + __expf(-(aU[r] + Iu + bu_)));
            const float rg = 1.0f / (1.0f + __expf(-(aR[r] + Ir + br_)));
            const float e  = tanhf(rg * aE[r] + Ie + be_);
            const float ph = prev_hidden[(size_t)row * HIDDEN + col];
            const float h  = u * ph + (1.0f - u) * e;
            hidden_out[(size_t)row * HIDDEN + col] = h;
            hnew_bf[(size_t)row * HIDDEN + col]    = f32_to_bf16(h);
        }
    };
    epi(aU0, aR0, aE0, rowBase);
    epi(aU1, aR1, aE1, rowBase + 16);
}

// ---------------------------------------------------------------------------
// Kernel 3: q = relu(h_half @ W_O{1,3}^T + b) -> bf16.
// Wave tile 32x32 (4 WMMAs / 8 loads), double buffered.
// grid = (SPLIT/32, BATCH/256, 2)  z: 0=coarse, 1=fine
// ---------------------------------------------------------------------------
__global__ __launch_bounds__(256) void mlp1_kernel(
    const unsigned short* __restrict__ hbf,   // bf16 [B,H]
    const unsigned short* __restrict__ W1, const float* __restrict__ b1,
    const unsigned short* __restrict__ W3, const float* __restrict__ b3,
    unsigned short* __restrict__ q1,          // bf16 [B,SPLIT]
    unsigned short* __restrict__ q2)
{
    const int lane    = threadIdx.x & 31;
    const int wave    = threadIdx.x >> 5;
    const int colBase = blockIdx.x * 32;
    const int rowBase = (blockIdx.y * 8 + wave) * 32;
    const bool fine   = (blockIdx.z != 0);

    const unsigned short* A    = hbf + (fine ? SPLIT : 0);  // column offset in [B,H]
    const unsigned short* W    = fine ? W3 : W1;
    const float*          bias = fine ? b3 : b1;
    unsigned short*       q    = fine ? q2 : q1;

    v8f c00 = {}, c01 = {}, c10 = {}, c11 = {};

    unsigned offA0 = a_boff(rowBase,      lane, HIDDEN);
    unsigned offA1 = a_boff(rowBase + 16, lane, HIDDEN);
    unsigned offB0 = b_boff(colBase,      lane, SPLIT);
    unsigned offB1 = b_boff(colBase + 16, lane, SPLIT);

    v16bf a0  = load_a_byte(A, offA0);
    v16bf a1  = load_a_byte(A, offA1);
    v16bf b0  = load_b_byte(W, offB0);
    v16bf b1f = load_b_byte(W, offB1);

    for (int it = 0; it < SPLIT / 32 - 1; ++it) {
        offA0 += 64; offA1 += 64; offB0 += 64; offB1 += 64;
        v16bf na0 = load_a_byte(A, offA0);
        v16bf na1 = load_a_byte(A, offA1);
        v16bf nb0 = load_b_byte(W, offB0);
        v16bf nb1 = load_b_byte(W, offB1);

        c00 = WMMA_BF16(a0, b0, c00);
        c01 = WMMA_BF16(a0, b1f, c01);
        c10 = WMMA_BF16(a1, b0, c10);
        c11 = WMMA_BF16(a1, b1f, c11);

        a0 = na0; a1 = na1; b0 = nb0; b1f = nb1;
    }
    c00 = WMMA_BF16(a0, b0, c00);
    c01 = WMMA_BF16(a0, b1f, c01);
    c10 = WMMA_BF16(a1, b0, c10);
    c11 = WMMA_BF16(a1, b1f, c11);

    const int cl0 = colBase + (lane & 15);
    const int cl1 = cl0 + 16;
    const float bb0 = bias[cl0];
    const float bb1 = bias[cl1];
    const int hi = (lane >> 4) ? 8 : 0;

    auto epi = [&](const v8f& c, int rb, int col, float bb) {
#pragma unroll
        for (int r = 0; r < 8; ++r) {
            const int row = rb + r + hi;
            float v = c[r] + bb;
            v = v > 0.0f ? v : 0.0f;
            q[(size_t)row * SPLIT + col] = f32_to_bf16(v);
        }
    };
    epi(c00, rowBase,      cl0, bb0);
    epi(c01, rowBase,      cl1, bb1);
    epi(c10, rowBase + 16, cl0, bb0);
    epi(c11, rowBase + 16, cl1, bb1);
}

// ---------------------------------------------------------------------------
// Kernel 4: out = q @ W_O{2,4}^T + b -> f32 into d_out.
// Wave tile 32x32, double buffered.
// grid = (QUANT/32, BATCH/256, 2)  z: 0=coarse, 1=fine
// ---------------------------------------------------------------------------
__global__ __launch_bounds__(256) void mlp2_kernel(
    const unsigned short* __restrict__ q1, const unsigned short* __restrict__ q2,
    const unsigned short* __restrict__ W2, const float* __restrict__ b2,
    const unsigned short* __restrict__ W4, const float* __restrict__ b4,
    float* __restrict__ out)                  // d_out base; fine at +B*QUANT
{
    const int lane    = threadIdx.x & 31;
    const int wave    = threadIdx.x >> 5;
    const int colBase = blockIdx.x * 32;
    const int rowBase = (blockIdx.y * 8 + wave) * 32;
    const bool fine   = (blockIdx.z != 0);

    const unsigned short* A    = fine ? q2 : q1;
    const unsigned short* W    = fine ? W4 : W2;
    const float*          bias = fine ? b4 : b2;
    float*                o    = out + (fine ? (size_t)BATCH * QUANT : 0);

    v8f c00 = {}, c01 = {}, c10 = {}, c11 = {};

    unsigned offA0 = a_boff(rowBase,      lane, SPLIT);
    unsigned offA1 = a_boff(rowBase + 16, lane, SPLIT);
    unsigned offB0 = b_boff(colBase,      lane, SPLIT);
    unsigned offB1 = b_boff(colBase + 16, lane, SPLIT);

    v16bf a0  = load_a_byte(A, offA0);
    v16bf a1  = load_a_byte(A, offA1);
    v16bf b0  = load_b_byte(W, offB0);
    v16bf b1f = load_b_byte(W, offB1);

    for (int it = 0; it < SPLIT / 32 - 1; ++it) {
        offA0 += 64; offA1 += 64; offB0 += 64; offB1 += 64;
        v16bf na0 = load_a_byte(A, offA0);
        v16bf na1 = load_a_byte(A, offA1);
        v16bf nb0 = load_b_byte(W, offB0);
        v16bf nb1 = load_b_byte(W, offB1);

        c00 = WMMA_BF16(a0, b0, c00);
        c01 = WMMA_BF16(a0, b1f, c01);
        c10 = WMMA_BF16(a1, b0, c10);
        c11 = WMMA_BF16(a1, b1f, c11);

        a0 = na0; a1 = na1; b0 = nb0; b1f = nb1;
    }
    c00 = WMMA_BF16(a0, b0, c00);
    c01 = WMMA_BF16(a0, b1f, c01);
    c10 = WMMA_BF16(a1, b0, c10);
    c11 = WMMA_BF16(a1, b1f, c11);

    const int cl0 = colBase + (lane & 15);
    const int cl1 = cl0 + 16;
    const float bb0 = bias[cl0];
    const float bb1 = bias[cl1];
    const int hi = (lane >> 4) ? 8 : 0;

    auto epi = [&](const v8f& c, int rb, int col, float bb) {
#pragma unroll
        for (int r = 0; r < 8; ++r) {
            const int row = rb + r + hi;
            o[(size_t)row * QUANT + col] = c[r] + bb;
        }
    };
    epi(c00, rowBase,      cl0, bb0);
    epi(c01, rowBase,      cl1, bb1);
    epi(c10, rowBase + 16, cl0, bb0);
    epi(c11, rowBase + 16, cl1, bb1);
}

// ---------------------------------------------------------------------------
// Host-side launch
// ---------------------------------------------------------------------------
extern "C" void kernel_launch(void* const* d_in, const int* in_sizes, int n_in,
                              void* d_out, int out_size, void* d_ws, size_t ws_size,
                              hipStream_t stream) {
    (void)in_sizes; (void)n_in; (void)out_size; (void)ws_size;

    const float* prev_y      = (const float*)d_in[0];
    const float* prev_hidden = (const float*)d_in[1];
    const float* cur_coarse  = (const float*)d_in[2];
    const float* W_Ru        = (const float*)d_in[3];
    const float* W_Rr        = (const float*)d_in[4];
    const float* W_Re        = (const float*)d_in[5];
    const float* W_Ic        = (const float*)d_in[6];
    const float* W_If        = (const float*)d_in[7];
    const float* W_O1        = (const float*)d_in[8];
    const float* b_O1        = (const float*)d_in[9];
    const float* W_O2        = (const float*)d_in[10];
    const float* b_O2        = (const float*)d_in[11];
    const float* W_O3        = (const float*)d_in[12];
    const float* b_O3        = (const float*)d_in[13];
    const float* W_O4        = (const float*)d_in[14];
    const float* b_O4        = (const float*)d_in[15];
    const float* bias_u      = (const float*)d_in[16];
    const float* bias_r      = (const float*)d_in[17];
    const float* bias_e      = (const float*)d_in[18];

    float* out_coarse = (float*)d_out;                                       // [B,QUANT]
    float* hidden_out = out_coarse + (size_t)2 * BATCH * QUANT;              // [B,H]

    // ---- workspace layout (bf16 elements) ----
    unsigned short* ws = (unsigned short*)d_ws;
    constexpr size_t N_HB  = (size_t)BATCH * HIDDEN;   // 7,340,032
    constexpr size_t N_WR  = (size_t)HIDDEN * HIDDEN;  //   802,816
    constexpr size_t N_W1  = (size_t)SPLIT * SPLIT;    //   200,704
    constexpr size_t N_W2  = (size_t)QUANT * SPLIT;    //   114,688
    constexpr size_t N_Q   = (size_t)BATCH * SPLIT;    // 3,670,016
    unsigned short* hprev_bf = ws;
    unsigned short* Wru_bf   = hprev_bf + N_HB;
    unsigned short* Wrr_bf   = Wru_bf + N_WR;
    unsigned short* Wre_bf   = Wrr_bf + N_WR;
    unsigned short* Wo1_bf   = Wre_bf + N_WR;
    unsigned short* Wo2_bf   = Wo1_bf + N_W1;
    unsigned short* Wo3_bf   = Wo2_bf + N_W2;
    unsigned short* Wo4_bf   = Wo3_bf + N_W1;
    unsigned short* hnew_bf  = Wo4_bf + N_W2;
    unsigned short* q1_bf    = hnew_bf + N_HB;
    unsigned short* q2_bf    = q1_bf + N_Q;

    // ---- 1) convert activations + weights to bf16 ----
    auto cvt = [&](const float* s, unsigned short* d, size_t n) {
        size_t n4 = n / 4;
        cvt_bf16_kernel<<<(unsigned)((n4 + 255) / 256), 256, 0, stream>>>(s, d, (int)n4);
    };
    cvt(prev_hidden, hprev_bf, N_HB);
    cvt(W_Ru, Wru_bf, N_WR);
    cvt(W_Rr, Wrr_bf, N_WR);
    cvt(W_Re, Wre_bf, N_WR);
    cvt(W_O1, Wo1_bf, N_W1);
    cvt(W_O2, Wo2_bf, N_W2);
    cvt(W_O3, Wo3_bf, N_W1);
    cvt(W_O4, Wo4_bf, N_W2);

    // ---- 2) fused recurrent GEMMs + gates ----
    gate_kernel<<<dim3(HIDDEN / 16, BATCH / 256), 256, 0, stream>>>(
        hprev_bf, Wru_bf, Wrr_bf, Wre_bf,
        prev_hidden, prev_y, cur_coarse, W_Ic, W_If,
        bias_u, bias_r, bias_e, hidden_out, hnew_bf);

    // ---- 3) hidden MLP layer 1 (relu) ----
    mlp1_kernel<<<dim3(SPLIT / 32, BATCH / 256, 2), 256, 0, stream>>>(
        hnew_bf, Wo1_bf, b_O1, Wo3_bf, b_O3, q1_bf, q2_bf);

    // ---- 4) output layer ----
    mlp2_kernel<<<dim3(QUANT / 32, BATCH / 256, 2), 256, 0, stream>>>(
        q1_bf, q2_bf, Wo2_bf, b_O2, Wo4_bf, b_O4, out_coarse);
}
